// RelationalLayerOriginal_37228776522176
// MI455X (gfx1250) — matrix operand order repaired and checked by reading, not compiled, not run on hardware
//
#include <hip/hip_runtime.h>

// ---------------------------------------------------------------------------
// Relation Network, fused for gfx1250 (wave32, WMMA 16x16x32 f16 -> f32 acc)
//
//   g-MLP (dominant, ~111 GFLOP): fused per 128-pair tile entirely in LDS.
//   Work split: each wave owns 2 N-tiles x all 128 rows -> every B fragment is
//   loaded from L2 exactly once per workgroup and stays in registers across
//   the 8 M-tile WMMAs (8x less L2 traffic than row-split).
//   f-MLP + log_softmax: tiny VALU kernel (<0.02% of FLOPs).
// ---------------------------------------------------------------------------

typedef __attribute__((ext_vector_type(16))) _Float16     v16h;
typedef __attribute__((ext_vector_type(8)))  float        v8f;
typedef __attribute__((ext_vector_type(4)))  unsigned int u32x4;

union Frag16 { v16h h; u32x4 u[2]; };

#define HIDN   256
#define MTILE  128     // pairs per workgroup
#define ASTR   264     // LDS row stride in halfs (264*2B = 33*16B: b128-aligned)

// packed-weight layout (halfs): frag(nt,ks) = 32 lanes x 16 halfs contiguous.
// lane L holds column n = nt*16 + L%16, K = ks*32 + (L/16)*16 + e  (e = 0..15)
#define OFF1 0                       // g1: 16 ntiles * 2 ksteps * 512
#define OFF2 16384                   // g2: 16 * 8 * 512
#define OFF3 (16384 + 65536)
#define OFF4 (16384 + 2 * 65536)
#define PACK_TOTAL (16384 + 3 * 65536)   // 212992 halfs = 425984 bytes

// ---------------------------------------------------------------------------
__global__ __launch_bounds__(256)
void pack_w_kernel(const float* __restrict__ W, int Kact, int N, int ksteps,
                   _Float16* __restrict__ dst) {
    const int total = 16 * ksteps * 512;
    for (int idx = blockIdx.x * blockDim.x + threadIdx.x; idx < total;
         idx += gridDim.x * blockDim.x) {
        const int e    = idx & 15;
        const int lane = (idx >> 4) & 31;
        const int ks   = (idx >> 9) % ksteps;
        const int nt   = idx / (ksteps << 9);
        const int n = nt * 16 + (lane & 15);
        const int k = ks * 32 + ((lane >> 4) << 4) + e;
        dst[idx] = (_Float16)((k < Kact) ? W[k * N + n] : 0.0f);
    }
}

__global__ void zerof_kernel(float* __restrict__ p, int n) {
    const int i = blockIdx.x * blockDim.x + threadIdx.x;
    if (i < n) p[i] = 0.0f;
}

// ---------------------------------------------------------------------------
// One 256-wide MLP layer over a 128-row LDS tile.
// 8 waves; wave w owns N-tiles {2w, 2w+1} across all 8 M-tiles.
__device__ __forceinline__
void mlp_layer(const _Float16* __restrict__ src, _Float16* __restrict__ dst,
               const _Float16* __restrict__ pw, const float* __restrict__ bias,
               int ksteps, int lane, int wave, bool reduce,
               float* __restrict__ xg_row) {
    const int half = lane >> 4;      // A: which 16-K run / D: which 8-row half
    const int col  = lane & 15;      // N within tile / M row within A tile

#pragma unroll
    for (int tt = 0; tt < 2; ++tt) {
        const int nt = wave * 2 + tt;
        v8f acc[8];
        const v8f vzero = {};
#pragma unroll
        for (int mt = 0; mt < 8; ++mt) acc[mt] = vzero;

        for (int ks = 0; ks < ksteps; ++ks) {
            // B (32x16 f16): 32 bytes/lane, pre-swizzled, contiguous; loaded
            // once and reused for all 8 M-tile WMMAs.
            Frag16 b;
            const _Float16* bp =
                pw + (((size_t)nt * ksteps + ks) * 32 + lane) * 16;
            b.u[0] = *(const u32x4*)(bp);
            b.u[1] = *(const u32x4*)(bp + 8);
#pragma unroll
            for (int mt = 0; mt < 8; ++mt) {
                // A (16x32 f16): one contiguous 16-half run per lane (K = 16*half + e)
                Frag16 a;
                const _Float16* ap =
                    src + (mt * 16 + col) * ASTR + ks * 32 + half * 16;
                a.u[0] = *(const u32x4*)(ap);
                a.u[1] = *(const u32x4*)(ap + 8);
                acc[mt] = __builtin_amdgcn_wmma_f32_16x16x32_f16(
                    false, a.h, false, b.h, (short)0, acc[mt], false, false);
            }
        }

        const float bn = bias[nt * 16 + col];
        if (!reduce) {
            // D layout: VGPR r, lane L -> row mt*16 + 8*(L/16) + r,
            //                            col nt*16 + L%16
#pragma unroll
            for (int mt = 0; mt < 8; ++mt) {
                _Float16* drow = dst + (mt * 16 + half * 8) * ASTR + nt * 16 + col;
#pragma unroll
                for (int r = 0; r < 8; ++r) {
                    float v = acc[mt][r] + bn;
                    drow[r * ASTR] = (_Float16)(v > 0.0f ? v : 0.0f);
                }
            }
        } else {
            // wave exclusively owns these columns: reduce 128 rows in-register,
            // fold the two lane-halves, one global atomic per column.
            float s = 0.0f;
#pragma unroll
            for (int mt = 0; mt < 8; ++mt)
#pragma unroll
                for (int r = 0; r < 8; ++r) {
                    float v = acc[mt][r] + bn;
                    s += (v > 0.0f ? v : 0.0f);
                }
            s += __shfl_xor(s, 16, 32);          // lane-permute, no LDS memory
            if (half == 0) atomicAdd(&xg_row[nt * 16 + col], s);
        }
    }
}

// ---------------------------------------------------------------------------
// Fused g-MLP. grid = 64 batches * 32 tiles = 2048 WGs of 256 threads.
__global__ __launch_bounds__(256)
void rn_g_kernel(const float* __restrict__ x,   const float* __restrict__ qst,
                 const float* __restrict__ g1b, const float* __restrict__ g2b,
                 const float* __restrict__ g3b, const float* __restrict__ g4b,
                 const _Float16* __restrict__ packed, float* __restrict__ xg) {
    extern __shared__ _Float16 smem[];
    _Float16* buf0 = smem;                 // [128][264] halfs
    _Float16* buf1 = smem + MTILE * ASTR;  // [128][264] halfs

    const int t    = threadIdx.x;
    const int lane = t & 31;
    const int wave = t >> 5;
    const int b     = blockIdx.x >> 5;
    const int pair0 = (blockIdx.x & 31) * MTILE;
    float* xg_row = xg + b * HIDN;

    // Build pair features [o_q(26) | o_p(26) | qst(11) | 0-pad] into buf0.
    // object i = y*8 + xx;  coords = linspace(-4, 4, 8) -> -4 + (8/7)*i
    for (int idx = t; idx < MTILE * 64; idx += 256) {
        const int row  = idx >> 6;
        const int cc   = idx & 63;
        const int pair = pair0 + row;
        const int p = pair >> 6, q = pair & 63;
        float v;
        if      (cc < 24)  v = x[((b * 24 + cc) << 6) + q];
        else if (cc == 24) v = -4.0f + (8.0f / 7.0f) * (float)(q & 7);
        else if (cc == 25) v = -4.0f + (8.0f / 7.0f) * (float)(q >> 3);
        else if (cc < 50)  v = x[((b * 24 + (cc - 26)) << 6) + p];
        else if (cc == 50) v = -4.0f + (8.0f / 7.0f) * (float)(p & 7);
        else if (cc == 51) v = -4.0f + (8.0f / 7.0f) * (float)(p >> 3);
        else if (cc < 63)  v = qst[b * 11 + (cc - 52)];
        else               v = 0.0f;
        buf0[row * ASTR + cc] = (_Float16)v;
    }
    __syncthreads();

    mlp_layer(buf0, buf1, packed + OFF1, g1b, 2, lane, wave, false, xg_row);
    __syncthreads();
    mlp_layer(buf1, buf0, packed + OFF2, g2b, 8, lane, wave, false, xg_row);
    __syncthreads();
    mlp_layer(buf0, buf1, packed + OFF3, g3b, 8, lane, wave, false, xg_row);
    __syncthreads();
    mlp_layer(buf1, buf0, packed + OFF4, g4b, 8, lane, wave, true,  xg_row);
}

// ---------------------------------------------------------------------------
// f-MLP + log_softmax. One block per batch row; negligible FLOPs.
__global__ __launch_bounds__(256)
void rn_f_kernel(const float* __restrict__ xg,
                 const float* __restrict__ f1w, const float* __restrict__ f1b,
                 const float* __restrict__ f2w, const float* __restrict__ f2b,
                 const float* __restrict__ f3w, const float* __restrict__ f3b,
                 float* __restrict__ out) {
    __shared__ float a[256], h[256], logits[28], red[2];
    const int b = blockIdx.x, t = threadIdx.x;

    a[t] = xg[b * 256 + t];
    __syncthreads();

    float acc = f1b[t];
    for (int k = 0; k < 256; ++k) acc = fmaf(a[k], f1w[k * 256 + t], acc);
    h[t] = fmaxf(acc, 0.0f);
    __syncthreads();

    acc = f2b[t];
    for (int k = 0; k < 256; ++k) acc = fmaf(h[k], f2w[k * 256 + t], acc);
    __syncthreads();
    a[t] = fmaxf(acc, 0.0f);
    __syncthreads();

    if (t < 28) {
        float l = f3b[t];
        for (int k = 0; k < 256; ++k) l = fmaf(a[k], f3w[k * 28 + t], l);
        logits[t] = l;
    }
    __syncthreads();
    if (t == 0) {
        float mx = logits[0];
        for (int i = 1; i < 28; ++i) mx = fmaxf(mx, logits[i]);
        float se = 0.0f;
        for (int i = 0; i < 28; ++i) se += __expf(logits[i] - mx);
        red[0] = mx;
        red[1] = __logf(se);
    }
    __syncthreads();
    if (t < 28) out[b * 28 + t] = logits[t] - red[0] - red[1];
}

// ---------------------------------------------------------------------------
extern "C" void kernel_launch(void* const* d_in, const int* in_sizes, int n_in,
                              void* d_out, int out_size, void* d_ws, size_t ws_size,
                              hipStream_t stream) {
    const float* x   = (const float*)d_in[0];
    const float* qst = (const float*)d_in[1];
    const float* g1w = (const float*)d_in[2];   const float* g1b = (const float*)d_in[3];
    const float* g2w = (const float*)d_in[4];   const float* g2b = (const float*)d_in[5];
    const float* g3w = (const float*)d_in[6];   const float* g3b = (const float*)d_in[7];
    const float* g4w = (const float*)d_in[8];   const float* g4b = (const float*)d_in[9];
    const float* f1w = (const float*)d_in[10];  const float* f1b = (const float*)d_in[11];
    const float* f2w = (const float*)d_in[12];  const float* f2b = (const float*)d_in[13];
    const float* f3w = (const float*)d_in[14];  const float* f3b = (const float*)d_in[15];
    float* out = (float*)d_out;

    // workspace: [packed f16 weights (425984 B)] [xg: 64*256 f32 (65536 B)]
    _Float16* packed = (_Float16*)d_ws;
    float*    xg     = (float*)((char*)d_ws + (size_t)PACK_TOTAL * 2);

    pack_w_kernel<<<64,  256, 0, stream>>>(g1w, 63,  256, 2, packed + OFF1);
    pack_w_kernel<<<256, 256, 0, stream>>>(g2w, 256, 256, 8, packed + OFF2);
    pack_w_kernel<<<256, 256, 0, stream>>>(g3w, 256, 256, 8, packed + OFF3);
    pack_w_kernel<<<256, 256, 0, stream>>>(g4w, 256, 256, 8, packed + OFF4);
    zerof_kernel<<<64, 256, 0, stream>>>(xg, 64 * 256);

    const size_t smem_bytes = (size_t)2 * MTILE * ASTR * sizeof(_Float16); // ~132 KB
    rn_g_kernel<<<2048, 256, smem_bytes, stream>>>(x, qst, g1b, g2b, g3b, g4b,
                                                   packed, xg);
    rn_f_kernel<<<64, 256, 0, stream>>>(xg, f1w, f1b, f2w, f2b, f3w, f3b, out);
}
